// MSAColumnGlobalAttention_36670430773503
// MI455X (gfx1250) — matrix operand order repaired and checked by reading, not compiled, not run on hardware
//
#include <hip/hip_runtime.h>

typedef __attribute__((ext_vector_type(16))) _Float16 v16h;
typedef __attribute__((ext_vector_type(8)))  _Float16 v8h;
typedef __attribute__((ext_vector_type(8)))  float    v8f;

#define B_DIM 2
#define S_DIM 512
#define L_DIM 384
#define CM    64

// ---- LDS layout (bytes), all offsets 32B aligned ----
#define XS_OFF     0        // f16 [512][72]
#define XS_STRIDE  72
#define LOGITS_OFF 73728    // f32 [8][520]
#define LG_STRIDE  520
#define ATTNA_OFF  90368    // f16 [16][520] (A-matrix for xa GEMM; rows 8..15 zero)
#define AT_STRIDE  520
#define MASKS_OFF  107008   // f32 [512]
#define ACC_OFF    109056   // f32 [64]
#define DEN_OFF    109312   // f32 [1]
#define QI_OFF     109344   // f32 [64]
#define QV_OFF     109600   // f32 [64]
#define QEB_OFF    109856   // f16 [16][64]  (B^T layout: [n][k])
#define WGT_OFF    111904   // f16 [64][64]  (wg transposed: [n][k])
#define WOGT_OFF   120096   // f16 [64][64]  (og*wo transposed: [n][k])
#define XA_OFF     128288   // f32 [8][64]
#define OG_OFF     130336   // f32 [64]
#define GSCR_OFF   130592   // f16 [8 waves][16][72]
#define SMEM_TOTAL 149024

__device__ __forceinline__ v8f wmma32(v16h a, v16h b, v8f c) {
  // D = A(16x32 f16) * B(32x16 f16) + C(16x16 f32)
  return __builtin_amdgcn_wmma_f32_16x16x32_f16(false, a, false, b, (short)0, c,
                                                false, false);
}

// A-matrix (16x32 f16) contiguous K order used here (matches load_B), not the ISA table order: lane holds row=lane&15;
// lanes 0-15: K = 0..15; lanes 16-31: K = 16..31.
__device__ __forceinline__ v16h load_A(const _Float16* base, int stride,
                                       int row0, int kbase, int lane) {
  int r  = row0 + (lane & 15);
  int k0 = kbase + ((lane < 16) ? 0 : 16);
  const _Float16* p = base + r * stride + k0;
  union { v16h v; v8h h[2]; } u;
  u.h[0] = *(const v8h*)p;          // K = k0 .. k0+7
  u.h[1] = *(const v8h*)(p + 8);    // K = k0+8 .. k0+15
  return u.v;
}

// B-matrix (32x16 f16): lane holds column n=lane&15;
// lanes 0-15: K=kbase..kbase+15, lanes 16-31: K=kbase+16..kbase+31.
// bt is stored N-major: bt[n][k], so the 16 K-halves are contiguous.
__device__ __forceinline__ v16h load_B(const _Float16* bt, int stride,
                                       int ntile, int kbase, int lane) {
  int n  = ntile * 16 + (lane & 15);
  int k0 = kbase + ((lane < 16) ? 0 : 16);
  return *(const v16h*)(bt + n * stride + k0);
}

// B-matrix gathered from a row-major [k][n] LDS array (K = sequence axis).
__device__ __forceinline__ v16h load_B_col(const _Float16* src, int stride,
                                           int ntile, int kbase, int lane) {
  int n  = ntile * 16 + (lane & 15);
  int k0 = kbase + ((lane < 16) ? 0 : 16);
  v16h b;
#pragma unroll
  for (int i = 0; i < 16; ++i) b[i] = src[(k0 + i) * stride + n];
  return b;
}

__global__ __launch_bounds__(256)
void MSAColumnGlobalAttention_kernel(
    const float* __restrict__ m,    const float* __restrict__ msk,
    const float* __restrict__ ln_w, const float* __restrict__ ln_b,
    const float* __restrict__ wq,   const float* __restrict__ wk,
    const float* __restrict__ wv,   const float* __restrict__ wg,
    const float* __restrict__ bg,   const float* __restrict__ wo,
    const float* __restrict__ bo,   float* __restrict__ out) {
  extern __shared__ char smem[];
  _Float16* xs    = (_Float16*)(smem + XS_OFF);
  float*    lgts  = (float*)(smem + LOGITS_OFF);
  _Float16* attnA = (_Float16*)(smem + ATTNA_OFF);
  float*    maskS = (float*)(smem + MASKS_OFF);
  float*    accv  = (float*)(smem + ACC_OFF);
  float*    denp  = (float*)(smem + DEN_OFF);
  float*    qi    = (float*)(smem + QI_OFF);
  float*    qv    = (float*)(smem + QV_OFF);
  _Float16* qeB   = (_Float16*)(smem + QEB_OFF);
  _Float16* wgT   = (_Float16*)(smem + WGT_OFF);
  _Float16* wogT  = (_Float16*)(smem + WOGT_OFF);
  float*    xa    = (float*)(smem + XA_OFF);
  float*    og    = (float*)(smem + OG_OFF);
  _Float16* gscr  = (_Float16*)(smem + GSCR_OFF);

  const int l    = blockIdx.x;   // residue index (0..383)
  const int bI   = blockIdx.y;   // batch (0..1)
  const int tid  = threadIdx.x;
  const int wave = tid >> 5;
  const int lane = tid & 31;

  // ---------------- P0: stage wg transposed to f16; zero accumulators ------
  for (int i = tid; i < 64 * 64; i += 256) {
    int k = i >> 6, n = i & 63;
    wgT[n * 64 + k] = (_Float16)wg[k * 64 + n];
  }
  for (int i = tid; i < 16 * AT_STRIDE; i += 256) attnA[i] = (_Float16)0.f;
  if (tid < 64) accv[tid] = 0.f;
  if (tid == 0) denp[0] = 0.f;
  __syncthreads();

  // ---------------- P1: LayerNorm + f16 stage + masked mean accumulation ---
  {
    float accx = 0.f, accy = 0.f, cnt = 0.f;
    const float2 lw = ((const float2*)ln_w)[lane];
    const float2 lb = ((const float2*)ln_b)[lane];
    for (int s = wave; s < S_DIM; s += 8) {
      const float* mr = m + (((size_t)bI * S_DIM + s) * L_DIM + l) * CM;
      __builtin_prefetch(mr + 8 * (size_t)L_DIM * CM, 0, 0);  // next s for this wave
      float2 v2 = ((const float2*)mr)[lane];
      float ssum = v2.x + v2.y;
      float ssq  = v2.x * v2.x + v2.y * v2.y;
#pragma unroll
      for (int o = 16; o > 0; o >>= 1) {
        ssum += __shfl_xor(ssum, o, 32);
        ssq  += __shfl_xor(ssq, o, 32);
      }
      float mu   = ssum * (1.f / 64.f);
      float var  = ssq * (1.f / 64.f) - mu * mu;
      float rinv = rsqrtf(var + 1e-5f);
      float x0 = (v2.x - mu) * rinv * lw.x + lb.x;
      float x1 = (v2.y - mu) * rinv * lw.y + lb.y;
      xs[s * XS_STRIDE + 2 * lane]     = (_Float16)x0;
      xs[s * XS_STRIDE + 2 * lane + 1] = (_Float16)x1;
      float mk = msk[((size_t)bI * S_DIM + s) * L_DIM + l];
      if (lane == 0) maskS[s] = mk;
      if (mk != 0.f) { accx += x0; accy += x1; cnt += 1.f; }
    }
    atomicAdd(&accv[2 * lane], accx);
    atomicAdd(&accv[2 * lane + 1], accy);
    if (lane == 0) atomicAdd(denp, cnt);
  }
  __syncthreads();

  // ---------------- P2: q_input, q = q_input@wq, fold wk@q -> qeB ----------
  if (tid < 64) {
    float d = fmaxf(denp[0], 1.f);
    qi[tid] = accv[tid] / d;
  }
  __syncthreads();
  if (tid < 64) {
    float s = 0.f;
    for (int mm = 0; mm < 64; ++mm) s += qi[mm] * wq[mm * 64 + tid];
    qv[tid] = s;
  }
  __syncthreads();
  for (int i = tid; i < 16 * 64; i += 256) {
    int n = i >> 6, k = i & 63;   // n = head (8..15 pad), k = feature
    float s = 0.f;
    if (n < 8) {
#pragma unroll
      for (int c = 0; c < 8; ++c) s += wk[k * 64 + n * 8 + c] * qv[n * 8 + c];
    }
    qeB[n * 64 + k] = (_Float16)s;
  }
  __syncthreads();

  // ---------------- P3: logits[h,s] = x @ wq_eff (WMMA), mask -------------
  {
    const float kscale = 0.35355339059327373f;  // 1/sqrt(8)
    v16h bq0 = load_B(qeB, 64, 0, 0, lane);
    v16h bq1 = load_B(qeB, 64, 0, 32, lane);
    for (int rb = wave; rb < 32; rb += 8) {
      v8f acc = {};
      acc = wmma32(load_A(xs, XS_STRIDE, rb * 16, 0, lane), bq0, acc);
      acc = wmma32(load_A(xs, XS_STRIDE, rb * 16, 32, lane), bq1, acc);
      int h = lane & 15;
      if (h < 8) {
#pragma unroll
        for (int r = 0; r < 8; ++r) {
          int s = rb * 16 + ((lane < 16) ? r : r + 8);
          float lg = acc[r] * kscale;
          lgts[h * LG_STRIDE + s] = (maskS[s] == 0.f) ? -1e9f : lg;
        }
      }
    }
  }
  __syncthreads();

  // ---------------- P4: softmax over S, one wave per head ------------------
  {
    int h = wave;  // 8 waves, 8 heads
    float mx = -3.4e38f;
    for (int s = lane; s < S_DIM; s += 32) mx = fmaxf(mx, lgts[h * LG_STRIDE + s]);
#pragma unroll
    for (int o = 16; o > 0; o >>= 1) mx = fmaxf(mx, __shfl_xor(mx, o, 32));
    float sm = 0.f;
    for (int s = lane; s < S_DIM; s += 32) sm += __expf(lgts[h * LG_STRIDE + s] - mx);
#pragma unroll
    for (int o = 16; o > 0; o >>= 1) sm += __shfl_xor(sm, o, 32);
    float inv = 1.f / sm;
    for (int s = lane; s < S_DIM; s += 32)
      attnA[h * AT_STRIDE + s] = (_Float16)(__expf(lgts[h * LG_STRIDE + s] - mx) * inv);
  }
  __syncthreads();

  // ---------------- P5: xa = attn(8x512) @ x(512x64) via WMMA, K along S ---
  if (wave < 4) {
    int nt = wave;
    v8f acc = {};
    for (int kc = 0; kc < 16; ++kc) {
      v16h a = load_A(attnA, AT_STRIDE, 0, kc * 32, lane);
      v16h b = load_B_col(xs, XS_STRIDE, nt, kc * 32, lane);
      acc = wmma32(a, b, acc);
    }
    if (lane < 16) {
#pragma unroll
      for (int r = 0; r < 8; ++r) xa[r * 64 + nt * 16 + lane] = acc[r];  // r = head
    }
  }
  __syncthreads();

  // ---------------- P6: out_global[h,c] = xa[h,:] . wv[:,8h+c] -------------
  if (tid < 64) {
    int h = tid >> 3, c = tid & 7;
    float s = 0.f;
    for (int mm = 0; mm < 64; ++mm) s += xa[h * 64 + mm] * wv[mm * 64 + h * 8 + c];
    og[tid] = s;
  }
  __syncthreads();

  // ---------------- P7: fold wog = diag(og) * wo (transposed, f16) ---------
  for (int i = tid; i < 64 * 64; i += 256) {
    int k = i >> 6, n = i & 63;
    wogT[n * 64 + k] = (_Float16)(og[k] * wo[k * 64 + n]);
  }
  __syncthreads();

  // ---------------- P8: out = sigmoid(x@wg+bg) @ wog + bo, masked ----------
  for (int rb = wave; rb < 32; rb += 8) {
    _Float16* gs = gscr + wave * 16 * XS_STRIDE;
    v16h a0 = load_A(xs, XS_STRIDE, rb * 16, 0, lane);
    v16h a1 = load_A(xs, XS_STRIDE, rb * 16, 32, lane);
#pragma unroll
    for (int nt = 0; nt < 4; ++nt) {
      v8f acc = {};
      acc = wmma32(a0, load_B(wgT, 64, nt, 0, lane), acc);
      acc = wmma32(a1, load_B(wgT, 64, nt, 32, lane), acc);
      int n = nt * 16 + (lane & 15);
      float bgn = bg[n];
#pragma unroll
      for (int r = 0; r < 8; ++r) {
        int Mr = (lane < 16) ? r : r + 8;
        float z = acc[r] + bgn;
        gs[Mr * XS_STRIDE + n] = (_Float16)(1.f / (1.f + __expf(-z)));
      }
    }
    // same-wave LDS ops are in-order; no workgroup barrier needed
    v16h ga0 = load_A(gs, XS_STRIDE, 0, 0, lane);
    v16h ga1 = load_A(gs, XS_STRIDE, 0, 32, lane);
#pragma unroll
    for (int nt = 0; nt < 4; ++nt) {
      v8f acc = {};
      acc = wmma32(ga0, load_B(wogT, 64, nt, 0, lane), acc);
      acc = wmma32(ga1, load_B(wogT, 64, nt, 32, lane), acc);
      int n = nt * 16 + (lane & 15);
      float bon = bo[n];
#pragma unroll
      for (int r = 0; r < 8; ++r) {
        int s = rb * 16 + ((lane < 16) ? r : r + 8);
        float val = (acc[r] + bon) * maskS[s];
        out[(((size_t)bI * S_DIM + s) * L_DIM + l) * CM + n] = val;
      }
    }
  }
}

extern "C" void kernel_launch(void* const* d_in, const int* in_sizes, int n_in,
                              void* d_out, int out_size, void* d_ws, size_t ws_size,
                              hipStream_t stream) {
  (void)in_sizes; (void)n_in; (void)out_size; (void)d_ws; (void)ws_size;
  const float* m    = (const float*)d_in[0];
  const float* msk  = (const float*)d_in[1];
  const float* ln_w = (const float*)d_in[2];
  const float* ln_b = (const float*)d_in[3];
  const float* wq   = (const float*)d_in[4];
  const float* wk   = (const float*)d_in[5];
  const float* wv   = (const float*)d_in[6];
  const float* wg   = (const float*)d_in[7];
  const float* bg   = (const float*)d_in[8];
  const float* wo   = (const float*)d_in[9];
  const float* bo   = (const float*)d_in[10];
  float* out = (float*)d_out;

  (void)hipFuncSetAttribute((const void*)MSAColumnGlobalAttention_kernel,
                            hipFuncAttributeMaxDynamicSharedMemorySize,
                            SMEM_TOTAL);
  dim3 grid(L_DIM, B_DIM);  // one workgroup per (l, b): 768 blocks
  MSAColumnGlobalAttention_kernel<<<grid, 256, SMEM_TOTAL, stream>>>(
      m, msk, ln_w, ln_b, wq, wk, wv, wg, bg, wo, bo, out);
}
